// HMHSA_24223615550110
// MI455X (gfx1250) — matrix operand order repaired and checked by reading, not compiled, not run on hardware
//
#include <hip/hip_runtime.h>

// ---------------------------------------------------------------------------
// Types for CDNA5 WMMA (wave32): bf16 16x16x32, f32 accumulate
// ---------------------------------------------------------------------------
typedef __attribute__((ext_vector_type(16))) __bf16 v16bf;
typedef __attribute__((ext_vector_type(8)))  __bf16 bf16x8;
typedef __attribute__((ext_vector_type(8)))  float  v8f;
typedef __attribute__((ext_vector_type(4)))  unsigned int u32x4;
typedef __attribute__((ext_vector_type(4)))  int i32x4;
typedef __attribute__((ext_vector_type(8)))  int i32x8;

#define DEV __device__ __forceinline__

// A fragment: 16(M) x 32(K) bf16.
// HW layout (16-bit A): lanes 0-15 hold K={0..7,16..23}, lanes 16-31 K={8..15,24..31}.
// We load logical K with the matching interleave so A/B logical-K slots pair up.
DEV v16bf lds_afrag(const __bf16* p, int ld, int rowbase, int k0, int lane) {
    int m = lane & 15, half = lane >> 4;
    const __bf16* r = p + (size_t)(rowbase + m) * ld + k0;
    bf16x8 c0 = *(const bf16x8*)(r + half * 8);        // hw K 0..7   / 8..15
    bf16x8 c1 = *(const bf16x8*)(r + 16 + half * 8);   // hw K 16..23 / 24..31
    v16bf out;
#pragma unroll
    for (int i = 0; i < 8; i++) { out[i] = c0[i]; out[8 + i] = c1[i]; }
    return out;
}

// B fragment: 32(K) x 16(N) bf16, operand stored as B'[n][k] row-major (column-of-B
// contiguous). HW layout: lanes 0-15 hold K={0..7,16..23} (N=lane), lanes 16-31 K={8..15,24..31}.
DEV v16bf lds_bfrag(const __bf16* p, int ld, int colbase, int k0, int lane) {
    int n = lane & 15, half = lane >> 4;
    const __bf16* r = p + (size_t)(colbase + n) * ld + k0;
    bf16x8 c0 = *(const bf16x8*)(r + half * 8);
    bf16x8 c1 = *(const bf16x8*)(r + 16 + half * 8);
    v16bf out;
#pragma unroll
    for (int i = 0; i < 8; i++) { out[i] = c0[i]; out[8 + i] = c1[i]; }
    return out;
}

DEV v8f wmma_bf16(v16bf a, v16bf b, v8f c) {
    return __builtin_amdgcn_wmma_f32_16x16x32_bf16(false, a, false, b, (short)0, c,
                                                   false, false);
}

// ---------------------------------------------------------------------------
// CDNA5 data movers
// ---------------------------------------------------------------------------
// Low 32 bits of a generic LDS pointer == workgroup-relative LDS byte address
// (LDS aperture rule: the address is truncated to its low 32 bits).
DEV unsigned lds_off(const void* p) {
    return (unsigned)(unsigned long long)p;
}

// Async global -> LDS copy of 16 bytes per lane (GLOBAL_LOAD_ASYNC_TO_LDS_B128,
// GV addressing: 64-bit VGPR address, SADDR=off). Tracked with ASYNCcnt.
DEV void async_b128(void* lds, const void* g) {
    asm volatile("global_load_async_to_lds_b128 %0, %1, off"
                 :
                 : "v"(lds_off(lds)), "v"((unsigned long long)g)
                 : "memory");
}
DEV void async_wait0() {
    asm volatile("s_wait_asynccnt 0x0" ::: "memory");
}

#if __has_builtin(__builtin_amdgcn_tensor_load_to_lds)
#define HAVE_TDM 1
// TDM load of a 2D bf16 tile: rows x 32 elements out of a row-major tensor with
// row stride 256 elements. Tile rows land densely in LDS (ld = 32 elements).
// D# layout per CDNA5 ISA Section 8.
DEV void tdm_load_2d(unsigned lds_addr, const void* gptr, int rows, int tdim1) {
    unsigned long long ga = (unsigned long long)gptr;
    u32x4 g0 = {
        1u,                                                   // count=1, user mode
        lds_addr,                                             // lds_addr (bytes)
        (unsigned)(ga & 0xffffffffu),                         // global_addr[31:0]
        (unsigned)((ga >> 32) & 0x01ffffffu) | (2u << 30)     // addr[56:32] | type=2
    };
    i32x8 g1 = {
        (int)0x00010000u,                                     // data_size = 2 bytes
        (int)(256u << 16),                                    // tensor_dim0 = 256
        (int)(((unsigned)tdim1 & 0xffffu) << 16),             // dim0_hi(0) | dim1_lo
        (int)((((unsigned)tdim1 >> 16) & 0xffffu) | (32u << 16)), // dim1_hi|tile_dim0
        rows,                                                 // tile_dim1 = rows
        256,                                                  // tensor_dim0_stride
        0, 0                                                  // dim1_stride unused
    };
    i32x4 z4 = {0, 0, 0, 0};
#if defined(__clang_major__) && (__clang_major__ >= 23)
    i32x8 z8 = {0, 0, 0, 0, 0, 0, 0, 0};
    __builtin_amdgcn_tensor_load_to_lds(g0, g1, z4, z4, z8, 0);
#else
    __builtin_amdgcn_tensor_load_to_lds(g0, g1, z4, z4, 0);
#endif
}
#endif

// ---------------------------------------------------------------------------
// Utility kernels
// ---------------------------------------------------------------------------
__global__ void k_zero_stats(float* st) {
    if (threadIdx.x < 24) st[threadIdx.x] = 0.0f;
}

__global__ void k_cvt_bf16(const float* __restrict__ src, __bf16* __restrict__ dst,
                           int count) {
    for (int i = blockIdx.x * blockDim.x + threadIdx.x; i < count;
         i += gridDim.x * blockDim.x)
        dst[i] = (__bf16)src[i];
}

// Per-sample sum / sumsq over contiguous [n][lenPerN] f32. st: [sum0,sum1,sq0,sq1,...]
__global__ void k_stats_partial(const float* __restrict__ in, float* st, int lenPerN) {
    int n = blockIdx.y;
    const float* p = in + (size_t)n * lenPerN;
    int tid = threadIdx.x;
    float s = 0.f, q = 0.f;
    for (int i = blockIdx.x * blockDim.x + tid; i < lenPerN;
         i += gridDim.x * blockDim.x) {
        float v = p[i];
        s += v; q += v * v;
    }
    __shared__ float rs[256], rq[256];
    rs[tid] = s; rq[tid] = q;
    __syncthreads();
    for (int off = 128; off; off >>= 1) {
        if (tid < off) { rs[tid] += rs[tid + off]; rq[tid] += rq[tid + off]; }
        __syncthreads();
    }
    if (tid == 0) { atomicAdd(&st[n], rs[0]); atomicAdd(&st[2 + n], rq[0]); }
}

__global__ void k_stats_final(float* st, int lenPerN) {
    int n = threadIdx.x;  // launched with 2 threads
    float inv = 1.0f / (float)lenPerN;
    float mean = st[n] * inv;
    float var = st[2 + n] * inv - mean * mean;
    st[4 + n] = mean;
    st[6 + n] = rsqrtf(var + 1e-6f);
}

// GroupNorm + transpose NCHWD f32 -> voxel-major [n*32768+v][256] bf16
__global__ void k_norm_x(const float* __restrict__ x, const float* __restrict__ st,
                         const float* __restrict__ w, const float* __restrict__ b,
                         __bf16* __restrict__ out) {
    size_t o = (size_t)blockIdx.x * 256 + threadIdx.x;
    int c = (int)(o & 255);
    size_t m = o >> 8;
    int n = (int)(m >> 15);
    int v = (int)(m & 32767);
    float val = (x[((size_t)n * 256 + c) * 32768 + v] - st[4 + n]) * st[6 + n] * w[c]
                + b[c];
    out[o] = (__bf16)val;
}

// GroupNorm elementwise on voxel-major contiguous f32 -> bf16
__global__ void k_norm_elem(const float* __restrict__ in, const float* __restrict__ st,
                            const float* __restrict__ w, const float* __restrict__ b,
                            __bf16* __restrict__ out, int lenPerN) {
    size_t o = (size_t)blockIdx.x * 256 + threadIdx.x;
    int c = (int)(o & 255);
    int n = (int)(o / (size_t)lenPerN);
    float val = (in[o] - st[4 + n]) * st[6 + n] * w[c] + b[c];
    out[o] = (__bf16)val;
}

// 4^3 average pool: attnx bf16 [n*32768+v][256] -> pooled f32 [n*512+s][256]
__global__ void k_pool(const __bf16* __restrict__ ax, float* __restrict__ pooled) {
    int blk = blockIdx.x;           // n*512 + s
    int n = blk >> 9, s = blk & 511;
    int ph = s >> 6, pw = (s >> 3) & 7, pd = s & 7;
    int c = threadIdx.x;
    float acc = 0.f;
#pragma unroll 4
    for (int ih = 0; ih < 4; ih++)
        for (int iw = 0; iw < 4; iw++)
            for (int id = 0; id < 4; id++) {
                int v = (4 * ph + ih) * 1024 + (4 * pw + iw) * 32 + (4 * pd + id);
                acc += (float)ax[((size_t)(n * 32768 + v)) * 256 + c];
            }
    pooled[(size_t)blk * 256 + c] = acc * (1.0f / 64.0f);
}

// ---------------------------------------------------------------------------
// Generic bf16 GEMM: out[M,Nc] = A[M,256] * W[Nc,256]^T + bias
// 256 threads = 8 waves (4x2), BM=128 BN=64 BK=32, WMMA bf16 16x16x32.
// Tiles staged into LDS by the Tensor Data Mover (wave 0 issues, TENSORcnt fenced).
// FINAL=true: write f32 transposed back to NCHWD (proj epilogue).
// ---------------------------------------------------------------------------
template <bool FINAL>
__global__ void k_gemm(const __bf16* __restrict__ A, const __bf16* __restrict__ W,
                       const float* __restrict__ bias, void* __restrict__ outp,
                       int M, int Nc) {
    __shared__ __bf16 sA[128 * 32];
    __shared__ __bf16 sW[64 * 32];
    int tid = threadIdx.x, lane = tid & 31, wid = tid >> 5;
    int m0 = blockIdx.x * 128, n0 = blockIdx.y * 64;
    int wr = wid >> 1, wc = wid & 1;
    v8f acc[2][2] = {};

#if defined(HAVE_TDM)
    unsigned ldsA = lds_off(sA);
    unsigned ldsW = lds_off(sW);
#endif

    for (int k0 = 0; k0 < 256; k0 += 32) {
        __syncthreads();
#if defined(HAVE_TDM)
        if (wid == 0) {   // TDM ignores EXEC; wave-uniform branch issues once
            tdm_load_2d(ldsA, A + (size_t)m0 * 256 + k0, 128, M);
            tdm_load_2d(ldsW, W + (size_t)n0 * 256 + k0, 64, Nc);
            __builtin_amdgcn_s_wait_tensorcnt(0);
        }
        if (k0 < 224)
            __builtin_prefetch(A + (size_t)(m0 + (tid >> 1)) * 256 + k0 + 32, 0, 0);
#else
        {
            int lr = tid >> 1, lh = tid & 1;
            const uint4* s = (const uint4*)(A + (size_t)(m0 + lr) * 256 + k0 + lh * 16);
            uint4* d = (uint4*)(sA + lr * 32 + lh * 16);
            d[0] = s[0]; d[1] = s[1];
            if (tid < 128) {
                const uint4* s2 =
                    (const uint4*)(W + (size_t)(n0 + lr) * 256 + k0 + lh * 16);
                uint4* d2 = (uint4*)(sW + lr * 32 + lh * 16);
                d2[0] = s2[0]; d2[1] = s2[1];
            }
        }
#endif
        __syncthreads();
#pragma unroll
        for (int rt = 0; rt < 2; rt++) {
            v16bf a = lds_afrag(sA, 32, wr * 32 + rt * 16, 0, lane);
#pragma unroll
            for (int ct = 0; ct < 2; ct++) {
                v16bf b = lds_bfrag(sW, 32, wc * 32 + ct * 16, 0, lane);
                acc[rt][ct] = wmma_bf16(a, b, acc[rt][ct]);
            }
        }
    }
    // epilogue
    int half = lane >> 4, nlo = lane & 15;
#pragma unroll
    for (int rt = 0; rt < 2; rt++)
#pragma unroll
        for (int ct = 0; ct < 2; ct++)
#pragma unroll
            for (int j = 0; j < 8; j++) {
                int mrow = m0 + wr * 32 + rt * 16 + j + 8 * half;
                int ncol = n0 + wc * 32 + ct * 16 + nlo;
                float v = acc[rt][ct][j] + bias[ncol];
                if (FINAL) {
                    float* out = (float*)outp;
                    int n = mrow >> 15, vox = mrow & 32767;
                    out[((size_t)n * 256 + ncol) * 32768 + vox] = v;
                } else {
                    __bf16* out = (__bf16*)outp;
                    out[(size_t)mrow * Nc + ncol] = (__bf16)v;
                }
            }
}

// ---------------------------------------------------------------------------
// Local windowed attention: one WG (4 waves) per (n, head, 4^3 window).
// S = (q/16) k^T  [64x64]; softmax; O = P v; y = x + O   (f32, voxel-major)
// q/k tiles staged with async global->LDS loads (ASYNCcnt fenced).
// ---------------------------------------------------------------------------
__global__ void k_local_attn(const __bf16* __restrict__ qkv,
                             const float* __restrict__ x, float* __restrict__ y) {
    __shared__ __bf16 sq[64 * 64], sk[64 * 64], svT[64 * 64], sP[64 * 64];
    __shared__ float sS[64 * 64];
    __shared__ int svox[64];
    __shared__ float srsum[64];

    int bid = blockIdx.x;
    int win = bid & 511, head = (bid >> 9) & 3, n = bid >> 11;
    int bh = win >> 6, bw = (win >> 3) & 7, bd = win & 7;
    int tid = threadIdx.x, lane = tid & 31, wid = tid >> 5;

    if (tid < 64) {
        int ih = tid >> 4, iw = (tid >> 2) & 3, id = tid & 3;
        svox[tid] = (bh * 4 + ih) * 1024 + (bw * 4 + iw) * 32 + (bd * 4 + id);
    }
    __syncthreads();

    {   // cooperative load of q,k (async, row-major) and v (transposed) tiles
        int r = tid >> 1, half = tid & 1;
        int v = svox[r];
        const __bf16* row = qkv + (size_t)(n * 32768 + v) * 768 + head * 64;
#pragma unroll
        for (int ch = 0; ch < 4; ch++) {
            async_b128(sq + r * 64 + half * 32 + ch * 8, row + half * 32 + ch * 8);
            async_b128(sk + r * 64 + half * 32 + ch * 8,
                       row + 256 + half * 32 + ch * 8);
        }
        {
            const __bf16* vr = row + 512 + half * 32;
            bf16x8 tv0 = *(const bf16x8*)(vr);
            bf16x8 tv1 = *(const bf16x8*)(vr + 8);
            bf16x8 tv2 = *(const bf16x8*)(vr + 16);
            bf16x8 tv3 = *(const bf16x8*)(vr + 24);
#pragma unroll
            for (int j = 0; j < 8; j++) {
                svT[(half * 32 + j) * 64 + r] = tv0[j];
                svT[(half * 32 + 8 + j) * 64 + r] = tv1[j];
                svT[(half * 32 + 16 + j) * 64 + r] = tv2[j];
                svT[(half * 32 + 24 + j) * 64 + r] = tv3[j];
            }
        }
        async_wait0();
    }
    __syncthreads();

    int m0 = wid * 16;
    int half = lane >> 4, nlo = lane & 15;

    // S = q k^T (scaled)
    {
        v8f acc[4] = {};
#pragma unroll
        for (int kk = 0; kk < 64; kk += 32) {
            v16bf a = lds_afrag(sq, 64, m0, kk, lane);
#pragma unroll
            for (int ct = 0; ct < 4; ct++) {
                v16bf b = lds_bfrag(sk, 64, ct * 16, kk, lane);
                acc[ct] = wmma_bf16(a, b, acc[ct]);
            }
        }
#pragma unroll
        for (int ct = 0; ct < 4; ct++)
#pragma unroll
            for (int j = 0; j < 8; j++)
                sS[(m0 + j + 8 * half) * 64 + ct * 16 + nlo] = acc[ct][j] * 0.0625f;
    }
    __syncthreads();

    // softmax: 2 lanes per row, 32 cols each
    {
        int row = m0 + (lane >> 1);
        int cb = (lane & 1) * 32;
        const float* sr = sS + row * 64 + cb;
        float mx = -3.0e38f;
#pragma unroll
        for (int j = 0; j < 32; j++) mx = fmaxf(mx, sr[j]);
        mx = fmaxf(mx, __shfl_xor(mx, 1, 32));
        float sum = 0.f;
#pragma unroll
        for (int j = 0; j < 32; j++) {
            float p = __expf(sr[j] - mx);
            sum += p;
            sP[row * 64 + cb + j] = (__bf16)p;
        }
        sum += __shfl_xor(sum, 1, 32);
        if ((lane & 1) == 0) srsum[row] = sum;
    }
    __syncthreads();

    // O = P v ; y = x + O / rowsum
    {
        v8f o[4] = {};
#pragma unroll
        for (int kk = 0; kk < 64; kk += 32) {
            v16bf a = lds_afrag(sP, 64, m0, kk, lane);
#pragma unroll
            for (int ct = 0; ct < 4; ct++) {
                v16bf b = lds_bfrag(svT, 64, ct * 16, kk, lane);
                o[ct] = wmma_bf16(a, b, o[ct]);
            }
        }
#pragma unroll
        for (int ct = 0; ct < 4; ct++)
#pragma unroll
            for (int j = 0; j < 8; j++) {
                int m = m0 + j + 8 * half;
                int col = ct * 16 + nlo;
                int v = svox[m];
                int c = head * 64 + col;
                float attn = o[ct][j] / srsum[m];
                float xres = x[((size_t)(n * 256 + c)) * 32768 + v];
                y[((size_t)(n * 32768 + v)) * 256 + c] = xres + attn;
            }
    }
}

// ---------------------------------------------------------------------------
// Global downsampled attention (flash-style): one WG per (n, head, 64-query block),
// 8 key-blocks of 64 from the 512 pooled tokens. g = softmax(q k^T / 16) v + attnx
// ---------------------------------------------------------------------------
__global__ void k_global_attn(const __bf16* __restrict__ q2,
                              const __bf16* __restrict__ kv2,
                              const __bf16* __restrict__ attnx,
                              __bf16* __restrict__ g) {
    __shared__ __bf16 sq[64 * 64], sk[64 * 64], svT[64 * 64], sP[64 * 64];
    __shared__ float sS[64 * 64];
    __shared__ float sm[64], sl[64], salpha[64];

    int bid = blockIdx.x;
    int qb = bid & 511, head = (bid >> 9) & 3, n = bid >> 11;
    int tid = threadIdx.x, lane = tid & 31, wid = tid >> 5;
    int lr = tid >> 1, lh = tid & 1;

    {   // q block (async copy)
        const __bf16* qrow = q2 + ((size_t)(n * 32768 + qb * 64 + lr)) * 256
                             + head * 64 + lh * 32;
#pragma unroll
        for (int ch = 0; ch < 4; ch++)
            async_b128(sq + lr * 64 + lh * 32 + ch * 8, qrow + ch * 8);
    }
    if (tid < 64) { sm[tid] = -3.0e38f; sl[tid] = 0.f; }

    int m0 = wid * 16;
    int half = lane >> 4, nlo = lane & 15;
    v8f o[4] = {};

    for (int kb = 0; kb < 8; kb++) {
        async_wait0();
        __syncthreads();
        {   // load k (async, row-major) and v (transposed) blocks
            const __bf16* row =
                kv2 + ((size_t)(n * 512 + kb * 64 + lr)) * 512 + head * 64;
#pragma unroll
            for (int ch = 0; ch < 4; ch++)
                async_b128(sk + lr * 64 + lh * 32 + ch * 8,
                           row + lh * 32 + ch * 8);
            {
                const __bf16* vr = row + 256 + lh * 32;
                bf16x8 tv0 = *(const bf16x8*)(vr);
                bf16x8 tv1 = *(const bf16x8*)(vr + 8);
                bf16x8 tv2 = *(const bf16x8*)(vr + 16);
                bf16x8 tv3 = *(const bf16x8*)(vr + 24);
#pragma unroll
                for (int j = 0; j < 8; j++) {
                    svT[(lh * 32 + j) * 64 + lr] = tv0[j];
                    svT[(lh * 32 + 8 + j) * 64 + lr] = tv1[j];
                    svT[(lh * 32 + 16 + j) * 64 + lr] = tv2[j];
                    svT[(lh * 32 + 24 + j) * 64 + lr] = tv3[j];
                }
            }
            async_wait0();
        }
        __syncthreads();

        // S block
        {
            v8f acc[4] = {};
#pragma unroll
            for (int kk = 0; kk < 64; kk += 32) {
                v16bf a = lds_afrag(sq, 64, m0, kk, lane);
#pragma unroll
                for (int ct = 0; ct < 4; ct++) {
                    v16bf b = lds_bfrag(sk, 64, ct * 16, kk, lane);
                    acc[ct] = wmma_bf16(a, b, acc[ct]);
                }
            }
#pragma unroll
            for (int ct = 0; ct < 4; ct++)
#pragma unroll
                for (int j = 0; j < 8; j++)
                    sS[(m0 + j + 8 * half) * 64 + ct * 16 + nlo] =
                        acc[ct][j] * 0.0625f;
        }
        __syncthreads();

        // online softmax update (per-wave rows)
        {
            int row = m0 + (lane >> 1);
            int cb = (lane & 1) * 32;
            const float* sr = sS + row * 64 + cb;
            float mold = sm[row];
            float bm = -3.0e38f;
#pragma unroll
            for (int j = 0; j < 32; j++) bm = fmaxf(bm, sr[j]);
            bm = fmaxf(bm, __shfl_xor(bm, 1, 32));
            float mnew = fmaxf(mold, bm);
            float sum = 0.f;
#pragma unroll
            for (int j = 0; j < 32; j++) {
                float p = __expf(sr[j] - mnew);
                sum += p;
                sP[row * 64 + cb + j] = (__bf16)p;
            }
            sum += __shfl_xor(sum, 1, 32);
            float alpha = __expf(mold - mnew);
            if ((lane & 1) == 0) {
                sm[row] = mnew;
                sl[row] = sl[row] * alpha + sum;
                salpha[row] = alpha;
            }
        }
        __syncthreads();

        // rescale accumulators, then O += P v
#pragma unroll
        for (int ct = 0; ct < 4; ct++)
#pragma unroll
            for (int j = 0; j < 8; j++)
                o[ct][j] *= salpha[m0 + j + 8 * half];
#pragma unroll
        for (int kk = 0; kk < 64; kk += 32) {
            v16bf a = lds_afrag(sP, 64, m0, kk, lane);
#pragma unroll
            for (int ct = 0; ct < 4; ct++) {
                v16bf b = lds_bfrag(svT, 64, ct * 16, kk, lane);
                o[ct] = wmma_bf16(a, b, o[ct]);
            }
        }
    }
    __syncthreads();

    // g = O/l + attnx residual
#pragma unroll
    for (int ct = 0; ct < 4; ct++)
#pragma unroll
        for (int j = 0; j < 8; j++) {
            int m = m0 + j + 8 * half;
            int col = ct * 16 + nlo;
            size_t ridx =
                ((size_t)(n * 32768 + qb * 64 + m)) * 256 + head * 64 + col;
            float val = o[ct][j] / sl[m] + (float)attnx[ridx];
            g[ridx] = (__bf16)val;
        }
}

// ---------------------------------------------------------------------------
// Launcher
// ---------------------------------------------------------------------------
extern "C" void kernel_launch(void* const* d_in, const int* in_sizes, int n_in,
                              void* d_out, int out_size, void* d_ws, size_t ws_size,
                              hipStream_t stream) {
    const float* x           = (const float*)d_in[0];
    const float* qkv_w       = (const float*)d_in[1];
    const float* qkv_b       = (const float*)d_in[2];
    const float* norm_w      = (const float*)d_in[3];
    const float* norm_b      = (const float*)d_in[4];
    const float* attn_norm_w = (const float*)d_in[5];
    const float* attn_norm_b = (const float*)d_in[6];
    const float* ds_norm_w   = (const float*)d_in[7];
    const float* ds_norm_b   = (const float*)d_in[8];
    const float* q_w         = (const float*)d_in[9];
    const float* q_b         = (const float*)d_in[10];
    const float* kv_w        = (const float*)d_in[11];
    const float* kv_b        = (const float*)d_in[12];
    const float* proj_w      = (const float*)d_in[13];
    const float* proj_b      = (const float*)d_in[14];

    char* ws = (char*)d_ws;
    size_t off = 0;
    float* st = (float*)(ws + off);       off += 1024;
    __bf16* wqkv = (__bf16*)(ws + off);   off += (size_t)768 * 256 * 2;
    __bf16* wq = (__bf16*)(ws + off);     off += (size_t)256 * 256 * 2;
    __bf16* wkv = (__bf16*)(ws + off);    off += (size_t)512 * 256 * 2;
    __bf16* wproj = (__bf16*)(ws + off);  off += (size_t)256 * 256 * 2;
    __bf16* xn = (__bf16*)(ws + off);     off += (size_t)65536 * 256 * 2;
    __bf16* qkv = (__bf16*)(ws + off);    off += (size_t)65536 * 768 * 2;
    float* y = (float*)(ws + off);        off += (size_t)65536 * 256 * 4;
    __bf16* attnx = (__bf16*)(ws + off);  off += (size_t)65536 * 256 * 2;
    float* pooled = (float*)(ws + off);   off += (size_t)1024 * 256 * 4;
    __bf16* pooledn = (__bf16*)(ws + off); off += (size_t)1024 * 256 * 2;
    __bf16* kv2 = (__bf16*)(ws + off);    off += (size_t)1024 * 512 * 2;
    __bf16* q2 = xn;   // xn dead after qkv GEMM
    __bf16* g = qkv;   // qkv dead after local attention

    k_zero_stats<<<1, 32, 0, stream>>>(st);
    k_cvt_bf16<<<192, 256, 0, stream>>>(qkv_w, wqkv, 768 * 256);
    k_cvt_bf16<<<64, 256, 0, stream>>>(q_w, wq, 256 * 256);
    k_cvt_bf16<<<128, 256, 0, stream>>>(kv_w, wkv, 512 * 256);
    k_cvt_bf16<<<64, 256, 0, stream>>>(proj_w, wproj, 256 * 256);

    // Stage 1: GroupNorm(x) -> qkv GEMM -> local window attention (+residual)
    k_stats_partial<<<dim3(512, 2), 256, 0, stream>>>(x, st + 0, 8388608);
    k_stats_final<<<1, 2, 0, stream>>>(st + 0, 8388608);
    k_norm_x<<<65536, 256, 0, stream>>>(x, st + 0, norm_w, norm_b, xn);
    k_gemm<false><<<dim3(512, 12), 256, 0, stream>>>(xn, wqkv, qkv_b, qkv, 65536, 768);
    k_local_attn<<<4096, 128, 0, stream>>>(qkv, x, y);

    // Stage 2: GroupNorm(x+attn) -> pool -> GroupNorm -> kv/q GEMMs
    k_stats_partial<<<dim3(512, 2), 256, 0, stream>>>(y, st + 8, 8388608);
    k_stats_final<<<1, 2, 0, stream>>>(st + 8, 8388608);
    k_norm_elem<<<65536, 256, 0, stream>>>(y, st + 8, attn_norm_w, attn_norm_b, attnx,
                                           8388608);
    k_pool<<<1024, 256, 0, stream>>>(attnx, pooled);
    k_stats_partial<<<dim3(64, 2), 256, 0, stream>>>(pooled, st + 16, 131072);
    k_stats_final<<<1, 2, 0, stream>>>(st + 16, 131072);
    k_norm_elem<<<1024, 256, 0, stream>>>(pooled, st + 16, ds_norm_w, ds_norm_b,
                                          pooledn, 131072);
    k_gemm<false><<<dim3(8, 8), 256, 0, stream>>>(pooledn, wkv, kv_b, kv2, 1024, 512);
    k_gemm<false><<<dim3(512, 4), 256, 0, stream>>>(attnx, wq, q_b, q2, 65536, 256);

    // Stage 3: global attention (+residual) -> proj GEMM (f32, NCHWD output)
    k_global_attn<<<4096, 128, 0, stream>>>(q2, kv2, attnx, g);
    k_gemm<true><<<dim3(512, 4), 256, 0, stream>>>(g, wproj, proj_b, d_out, 65536, 256);
}
